// BiologicalMultiHeadAttention_37538014167119
// MI455X (gfx1250) — matrix operand order repaired and checked by reading, not compiled, not run on hardware
//
#include <hip/hip_runtime.h>

// Problem constants (from reference)
#define B_  2
#define S_  2048
#define E_  1024
#define H_  16
#define D_  64
#define KK_ 409            // max(1, int(S*0.2))

// fused score scale: (1/sqrt(D)) / temperature = (1/8) * 1.25
#define SC_SCALE 0.15625f
#define DIAG_MUL 1.15f     // 1 + ACH*0.3
#define TOPK_MUL 1.15f     // 1 + (-DA)*0.3

typedef __attribute__((ext_vector_type(16))) _Float16 v16h;
typedef __attribute__((ext_vector_type(8)))  _Float16 v8h;
typedef __attribute__((ext_vector_type(4)))  _Float16 v4h;
typedef __attribute__((ext_vector_type(8)))  float    v8f;
typedef __attribute__((ext_vector_type(4)))  float    v4f;

__device__ __forceinline__ v8f wmma16(v16h a, v16h b, v8f c) {
  return __builtin_amdgcn_wmma_f32_16x16x32_f16(false, a, false, b, (short)0, c,
                                                false, false);
}

// A-matrix fragment (16x32 f16). rowptr = this lane's row (row-major source).
// ISA: lanes 0-15 -> K = {e, 16+e}; lanes 16-31 -> K = {8+e, 24+e}.
__device__ __forceinline__ v16h fragA_rows(const _Float16* rowptr, int k0, int lane) {
  const int base = (lane & 16) ? 8 : 0;
  const v8h lo = *(const v8h*)(rowptr + k0 + base);
  const v8h hi = *(const v8h*)(rowptr + k0 + 16 + base);
  return __builtin_shufflevector(lo, hi, 0, 1, 2, 3, 4, 5, 6, 7,
                                 8, 9, 10, 11, 12, 13, 14, 15);
}

// B-matrix fragment (32x16 f16): lanes 0-15 hold K={0..7,16..23} of column (lane&15),
// lanes 16-31 hold K={8..15,24..31}. colptr = K=0 of this lane's column, K contiguous.
__device__ __forceinline__ v16h fragB_cols(const _Float16* colptr, int lane) {
  const int kofs = (lane & 16) ? 8 : 0;
  const v8h lo = *(const v8h*)(colptr + kofs);
  const v8h hi = *(const v8h*)(colptr + kofs + 16);
  return __builtin_shufflevector(lo, hi, 0, 1, 2, 3, 4, 5, 6, 7,
                                 8, 9, 10, 11, 12, 13, 14, 15);
}

__device__ __forceinline__ unsigned f2ord(float f) {
  unsigned u = __float_as_uint(f);
  return (u & 0x80000000u) ? ~u : (u | 0x80000000u);
}

// ---------------------------------------------------------------------------
// GEMM tiling: 128x128 block, BK=32, 256 threads (8 wave32s), wave = 32x64
// sub-tile (2x4 WMMA tiles). Register-staged DOUBLE-BUFFERED pipeline:
// next tile's global loads are issued before this tile's 8 WMMAs, hiding
// HBM latency behind matrix math; one barrier per K-step.
// ---------------------------------------------------------------------------
#define AS_STRIDE 40    // 128 x 32 f16 A tile (row-major), padded
#define BT_STRIDE 40    // 128(n) x 32(k) f16 B tile (transposed), padded
#define NKT (E_ / 32)   // 32 K-steps

// Projection GEMM: Y = (X(fp32) * W + bias) [* hscale[head]], f16 out in
// [b,h,s,d] layout. HAS_SCALE selects the time_scales path at compile time.
template <bool HAS_SCALE>
__global__ __launch_bounds__(256)
void proj_gemm_kernel(const float* __restrict__ X,      // [B*S, E]
                      const float* __restrict__ W,      // [E, E]
                      const float* __restrict__ bias,   // [E]
                      const float* __restrict__ hscale, // [H]
                      _Float16* __restrict__ Y)         // [B*H, S, D]
{
  __shared__ _Float16 As [2][128 * AS_STRIDE];
  __shared__ _Float16 BsT[2][128 * BT_STRIDE];
  const int tid = threadIdx.x, lane = tid & 31, wave = tid >> 5;
  const int n0 = blockIdx.x * 128, m0 = blockIdx.y * 128;
  const int wm = (wave >> 1) * 32, wn = (wave & 1) * 64;

  v8f acc[2][4];
  const v8f vz = {};
#pragma unroll
  for (int i = 0; i < 2; ++i)
#pragma unroll
    for (int j = 0; j < 4; ++j) acc[i][j] = vz;

  float4 xa[4];     // staged X tile (this thread's share)
  float  wb[4][4];  // staged W tile

  auto gload = [&](int k0) {
#pragma unroll
    for (int p = 0; p < 4; ++p) {
      const int r = p * 32 + (tid >> 3), c = (tid & 7) * 4;
      xa[p] = *(const float4*)(X + (size_t)(m0 + r) * E_ + k0 + c);
    }
#pragma unroll
    for (int t = 0; t < 4; ++t) {
      const int tsk = tid + t * 256;
      const int c = tsk & 127, rg = (tsk >> 7) * 4;  // n (coalesced), k group
#pragma unroll
      for (int u = 0; u < 4; ++u)
        wb[t][u] = W[(size_t)(k0 + rg + u) * E_ + n0 + c];
    }
  };
  auto lstore = [&](int bsel) {
#pragma unroll
    for (int p = 0; p < 4; ++p) {
      const int r = p * 32 + (tid >> 3), c = (tid & 7) * 4;
      v4h t; t[0] = (_Float16)xa[p].x; t[1] = (_Float16)xa[p].y;
             t[2] = (_Float16)xa[p].z; t[3] = (_Float16)xa[p].w;
      *(v4h*)(As[bsel] + r * AS_STRIDE + c) = t;
    }
#pragma unroll
    for (int t = 0; t < 4; ++t) {
      const int tsk = tid + t * 256;
      const int c = tsk & 127, rg = (tsk >> 7) * 4;
      v4h tw;
#pragma unroll
      for (int u = 0; u < 4; ++u) tw[u] = (_Float16)wb[t][u];
      *(v4h*)(BsT[bsel] + c * BT_STRIDE + rg) = tw;
    }
  };
  auto compute = [&](int bsel) {
    v16h a[2];
#pragma unroll
    for (int i = 0; i < 2; ++i)
      a[i] = fragA_rows(As[bsel] + (wm + i * 16 + (lane & 15)) * AS_STRIDE, 0, lane);
#pragma unroll
    for (int j = 0; j < 4; ++j) {
      v16h b = fragB_cols(BsT[bsel] + (wn + j * 16 + (lane & 15)) * BT_STRIDE, lane);
#pragma unroll
      for (int i = 0; i < 2; ++i) acc[i][j] = wmma16(a[i], b, acc[i][j]);
    }
  };

  gload(0);
  lstore(0);
  __syncthreads();
#pragma unroll 1
  for (int kt = 0; kt < NKT; ++kt) {
    if (kt + 1 < NKT) gload((kt + 1) * 32);   // in flight during the WMMAs
    compute(kt & 1);
    if (kt + 1 < NKT) lstore((kt + 1) & 1);
    __syncthreads();
  }

  // Epilogue: bias/scale per-j; HAS_SCALE constant-folds the branch away.
#pragma unroll
  for (int j = 0; j < 4; ++j) {
    const int ncol = n0 + wn + j * 16 + (lane & 15);
    const int h = ncol >> 6, d = ncol & 63;
    const float badd = bias[ncol];
    const float scl = HAS_SCALE ? hscale[h] : 1.0f;  // time_scales on K proj
    _Float16* yh = Y + ((size_t)h * S_) * D_ + d;
#pragma unroll
    for (int i = 0; i < 2; ++i)
#pragma unroll
      for (int r = 0; r < 8; ++r) {
        const int mrow = m0 + wm + i * 16 + r + ((lane & 16) ? 8 : 0);
        const int bb = mrow >> 11, s = mrow & (S_ - 1);
        const float v = (acc[i][j][r] + badd) * scl;
        yh[((size_t)bb * H_ * S_ + s) * D_] = (_Float16)v;
      }
  }
}

// Output GEMM: d_out = AttnOut(f16) * Wo + bo, fp32 result. Same pipeline.
__global__ __launch_bounds__(256)
void out_gemm_kernel(const _Float16* __restrict__ Xh,  // [B*S, E] f16
                     const float* __restrict__ W,      // Wo [E, E]
                     const float* __restrict__ bias,   // bo [E]
                     float* __restrict__ Y)            // [B*S, E] fp32
{
  __shared__ _Float16 As [2][128 * AS_STRIDE];
  __shared__ _Float16 BsT[2][128 * BT_STRIDE];
  const int tid = threadIdx.x, lane = tid & 31, wave = tid >> 5;
  const int n0 = blockIdx.x * 128, m0 = blockIdx.y * 128;
  const int wm = (wave >> 1) * 32, wn = (wave & 1) * 64;

  v8f acc[2][4];
  const v8f vz = {};
#pragma unroll
  for (int i = 0; i < 2; ++i)
#pragma unroll
    for (int j = 0; j < 4; ++j) acc[i][j] = vz;

  v8h   xa[2];      // staged A tile (f16 passthrough)
  float wb[4][4];   // staged Wo tile

  auto gload = [&](int k0) {
#pragma unroll
    for (int t = 0; t < 2; ++t) {
      const int c = tid + t * 256;               // 512 16B chunks
      const int r = c >> 2, co = (c & 3) * 8;
      xa[t] = *(const v8h*)(Xh + (size_t)(m0 + r) * E_ + k0 + co);
    }
#pragma unroll
    for (int t = 0; t < 4; ++t) {
      const int tsk = tid + t * 256;
      const int c = tsk & 127, rg = (tsk >> 7) * 4;
#pragma unroll
      for (int u = 0; u < 4; ++u)
        wb[t][u] = W[(size_t)(k0 + rg + u) * E_ + n0 + c];
    }
  };
  auto lstore = [&](int bsel) {
#pragma unroll
    for (int t = 0; t < 2; ++t) {
      const int c = tid + t * 256;
      const int r = c >> 2, co = (c & 3) * 8;
      *(v8h*)(As[bsel] + r * AS_STRIDE + co) = xa[t];
    }
#pragma unroll
    for (int t = 0; t < 4; ++t) {
      const int tsk = tid + t * 256;
      const int c = tsk & 127, rg = (tsk >> 7) * 4;
      v4h tw;
#pragma unroll
      for (int u = 0; u < 4; ++u) tw[u] = (_Float16)wb[t][u];
      *(v4h*)(BsT[bsel] + c * BT_STRIDE + rg) = tw;
    }
  };
  auto compute = [&](int bsel) {
    v16h a[2];
#pragma unroll
    for (int i = 0; i < 2; ++i)
      a[i] = fragA_rows(As[bsel] + (wm + i * 16 + (lane & 15)) * AS_STRIDE, 0, lane);
#pragma unroll
    for (int j = 0; j < 4; ++j) {
      v16h b = fragB_cols(BsT[bsel] + (wn + j * 16 + (lane & 15)) * BT_STRIDE, lane);
#pragma unroll
      for (int i = 0; i < 2; ++i) acc[i][j] = wmma16(a[i], b, acc[i][j]);
    }
  };

  gload(0);
  lstore(0);
  __syncthreads();
#pragma unroll 1
  for (int kt = 0; kt < NKT; ++kt) {
    if (kt + 1 < NKT) gload((kt + 1) * 32);
    compute(kt & 1);
    if (kt + 1 < NKT) lstore((kt + 1) & 1);
    __syncthreads();
  }

#pragma unroll
  for (int j = 0; j < 4; ++j) {
    const int ncol = n0 + wn + j * 16 + (lane & 15);
    const float badd = bias[ncol];
#pragma unroll
    for (int i = 0; i < 2; ++i)
#pragma unroll
      for (int r = 0; r < 8; ++r) {
        const int mrow = m0 + wm + i * 16 + r + ((lane & 16) ? 8 : 0);
        Y[(size_t)mrow * E_ + ncol] = acc[i][j][r] + badd;
      }
  }
}

// ---------------------------------------------------------------------------
// V transpose: [bh, s, d] -> [bh, d, s] so P*V B-fragments are contiguous-in-K.
// ---------------------------------------------------------------------------
__global__ __launch_bounds__(256)
void transpose_v_kernel(const _Float16* __restrict__ v,  // [BH, S, D]
                        _Float16* __restrict__ vt)       // [BH, D, S]
{
  __shared__ _Float16 tile[64 * 80];   // 64(s) x 64(d), padded stride 80
  const int tid = threadIdx.x;
  const int bh = blockIdx.y;
  const int s0 = blockIdx.x * 64;
  const _Float16* src = v + ((size_t)bh * S_ + s0) * D_;
  {
    const int r = tid >> 2;            // s row 0..63
    const int co = (tid & 3) * 16;     // d chunk
    *(v8h*)(tile + r * 80 + co)     = *(const v8h*)(src + r * D_ + co);
    *(v8h*)(tile + r * 80 + co + 8) = *(const v8h*)(src + r * D_ + co + 8);
  }
  __syncthreads();
  {
    const int d = tid >> 2;            // d row 0..63
    const int sc = (tid & 3) * 16;     // s chunk
    v8h o0, o1;
#pragma unroll
    for (int u = 0; u < 8; ++u) {
      o0[u] = tile[(sc + u) * 80 + d];
      o1[u] = tile[(sc + 8 + u) * 80 + d];
    }
    _Float16* dst = vt + ((size_t)bh * D_ + d) * S_ + s0 + sc;
    *(v8h*)(dst)     = o0;
    *(v8h*)(dst + 8) = o1;
  }
}

// ---------------------------------------------------------------------------
// Attention: one workgroup = (b,h) x 32 query rows; 32x2048 fp32 score panel
// resident in LDS (~271 KB of the 320 KB WGP LDS). Exact top-k via 32-step
// radix descent with the row's 64 ordered keys cached in VGPRs.
// 8 waves; each owns a 256-key slice for both Q*K^T and P*V.
// ---------------------------------------------------------------------------
#define SC_STRIDE 2052   // 2048 + 4 floats pad
#define OA_STRIDE 68

__global__ __launch_bounds__(256)
void attn_kernel(const _Float16* __restrict__ q,   // [B*H, S, D]
                 const _Float16* __restrict__ k,   // [B*H, S, D] (pre-scaled)
                 const _Float16* __restrict__ vt,  // [B*H, D, S] (transposed)
                 _Float16* __restrict__ ao)        // [B, S, E]
{
  extern __shared__ char smem_raw[];
  float* sc     = (float*)smem_raw;                // 32 x SC_STRIDE
  float* outAcc = sc + 32 * SC_STRIDE;             // 32 x OA_STRIDE
  float* invSum = outAcc + 32 * OA_STRIDE;         // 32

  const int tid = threadIdx.x, lane = tid & 31, wave = tid >> 5;
  const int bh = blockIdx.x >> 6;                  // b*H + h
  const int m0 = (blockIdx.x & 63) * 32;           // query row block
  const _Float16* qh = q  + (size_t)bh * S_ * D_;
  const _Float16* kh = k  + (size_t)bh * S_ * D_;
  const _Float16* vh = vt + (size_t)bh * D_ * S_;
  const int slice0 = wave * 256;                   // this wave's key slice

  // ---- phase 1: scores = (q k^T) * SC_SCALE, diagonal boost, into LDS ----
  v16h aq[2][2];
#pragma unroll
  for (int i = 0; i < 2; ++i)
#pragma unroll
    for (int ks = 0; ks < 2; ++ks)
      aq[i][ks] = fragA_rows(qh + (size_t)(m0 + i * 16 + (lane & 15)) * D_,
                             ks * 32, lane);
  for (int j = 0; j < 16; ++j) {
    const int colL = slice0 + j * 16 + (lane & 15);
    v16h b0 = fragB_cols(kh + (size_t)colL * D_ + 0,  lane);
    v16h b1 = fragB_cols(kh + (size_t)colL * D_ + 32, lane);
#pragma unroll
    for (int i = 0; i < 2; ++i) {
      v8f acc = {};
      acc = wmma16(aq[i][0], b0, acc);
      acc = wmma16(aq[i][1], b1, acc);
#pragma unroll
      for (int r = 0; r < 8; ++r) {
        const int qrow = i * 16 + r + ((lane & 16) ? 8 : 0);
        const int key  = slice0 + j * 16 + (lane & 15);
        float s = acc[r] * SC_SCALE;
        if (m0 + qrow == key) s *= DIAG_MUL;
        sc[qrow * SC_STRIDE + key] = s;
      }
    }
  }
  for (int idx = tid; idx < 32 * OA_STRIDE; idx += 256) outAcc[idx] = 0.0f;
  __syncthreads();

  // ---- phase 2: per-row exact k-th largest + mask + softmax (wave/row) ----
  for (int rr = 0; rr < 4; ++rr) {
    const int row = wave + rr * 8;
    float* srow = sc + row * SC_STRIDE;

    unsigned keys[64];                     // this lane's 64 keys, in VGPRs
#pragma unroll
    for (int t = 0; t < 64; ++t) keys[t] = f2ord(srow[lane + t * 32]);

    unsigned prefix = 0u;
    int kneed = KK_;
#pragma unroll 1
    for (int bit = 31; bit >= 0; --bit) {  // radix descent, exact kth largest
      const unsigned want = (prefix >> bit) | 1u;
      int cnt = 0;
#pragma unroll
      for (int t = 0; t < 64; ++t) cnt += (int)((keys[t] >> bit) == want);
#pragma unroll
      for (int off = 16; off > 0; off >>= 1) cnt += __shfl_down(cnt, off, 32);
      cnt = __shfl(cnt, 0, 32);
      if (cnt >= kneed) prefix |= (1u << bit);
      else kneed -= cnt;
    }
    const unsigned thr = prefix;           // ordered bits of min_topk

    float rmax = -3.0e38f;
#pragma unroll
    for (int t = 0; t < 64; ++t) {         // mask boost + row max
      float s = srow[lane + t * 32];
      if (keys[t] >= thr) s *= TOPK_MUL;   // scores >= min_topk
      srow[lane + t * 32] = s;
      rmax = fmaxf(rmax, s);
    }
#pragma unroll
    for (int off = 16; off > 0; off >>= 1)
      rmax = fmaxf(rmax, __shfl_down(rmax, off, 32));
    rmax = __shfl(rmax, 0, 32);

    float rsum = 0.0f;
#pragma unroll
    for (int t = 0; t < 64; ++t) {
      const float e = __expf(srow[lane + t * 32] - rmax);
      srow[lane + t * 32] = e;
      rsum += e;
    }
#pragma unroll
    for (int off = 16; off > 0; off >>= 1) rsum += __shfl_down(rsum, off, 32);
    if (lane == 0) invSum[row] = 1.0f / rsum;
  }
  __syncthreads();

  // ---- phase 3: out = P * V, per-wave partial over its 256-key slice ----
  v8f oacc[2][4];
  const v8f vz = {};
#pragma unroll
  for (int i = 0; i < 2; ++i)
#pragma unroll
    for (int j = 0; j < 4; ++j) oacc[i][j] = vz;

  for (int ks = 0; ks < 8; ++ks) {
    const int kb = slice0 + ks * 32;
    v16h a[2];
#pragma unroll
    for (int i = 0; i < 2; ++i) {          // A frag from LDS probs, b128 loads
      const float* srow = sc + (i * 16 + (lane & 15)) * SC_STRIDE;
      const int base = (lane & 16) ? 8 : 0;
      const v4f x0 = *(const v4f*)(srow + kb + base);
      const v4f x1 = *(const v4f*)(srow + kb + base + 4);
      const v4f x2 = *(const v4f*)(srow + kb + 16 + base);
      const v4f x3 = *(const v4f*)(srow + kb + 16 + base + 4);
      v16h f;
#pragma unroll
      for (int e = 0; e < 4; ++e) {
        f[e]      = (_Float16)x0[e];
        f[4 + e]  = (_Float16)x1[e];
        f[8 + e]  = (_Float16)x2[e];
        f[12 + e] = (_Float16)x3[e];
      }
      a[i] = f;
    }
#pragma unroll
    for (int j = 0; j < 4; ++j) {
      const int d = j * 16 + (lane & 15);
      v16h bf = fragB_cols(vh + (size_t)d * S_ + kb, lane);
#pragma unroll
      for (int i = 0; i < 2; ++i) oacc[i][j] = wmma16(a[i], bf, oacc[i][j]);
    }
  }
#pragma unroll
  for (int i = 0; i < 2; ++i)              // cross-wave reduce via ds_add_f32
#pragma unroll
    for (int j = 0; j < 4; ++j)
#pragma unroll
      for (int r = 0; r < 8; ++r) {
        const int row = i * 16 + r + ((lane & 16) ? 8 : 0);
        const int d   = j * 16 + (lane & 15);
        atomicAdd(&outAcc[row * OA_STRIDE + d], oacc[i][j][r]);
      }
  __syncthreads();

  const int bb = bh >> 4, h = bh & 15;
  {                                        // vectorized 16B output stores
    const int row = tid >> 3, dco = (tid & 7) * 8;
    const float is = invSum[row];
    v8h o;
#pragma unroll
    for (int u = 0; u < 8; ++u)
      o[u] = (_Float16)(outAcc[row * OA_STRIDE + dco + u] * is);
    *(v8h*)(ao + ((size_t)bb * S_ + (m0 + row)) * E_ + h * D_ + dco) = o;
  }
}

// ---------------------------------------------------------------------------
extern "C" void kernel_launch(void* const* d_in, const int* in_sizes, int n_in,
                              void* d_out, int out_size, void* d_ws, size_t ws_size,
                              hipStream_t stream) {
  const float* query = (const float*)d_in[0];
  const float* key   = (const float*)d_in[1];
  const float* value = (const float*)d_in[2];
  const float* Wq = (const float*)d_in[3];
  const float* bq = (const float*)d_in[4];
  const float* Wk = (const float*)d_in[5];
  const float* bk = (const float*)d_in[6];
  const float* Wv = (const float*)d_in[7];
  const float* bv = (const float*)d_in[8];
  const float* Wo = (const float*)d_in[9];
  const float* bo = (const float*)d_in[10];
  const float* ts = (const float*)d_in[11];
  float* out = (float*)d_out;

  const size_t NTOK = (size_t)B_ * S_;      // 4096 rows
  _Float16* qb = (_Float16*)d_ws;           // 8 MB each, 40 MB total
  _Float16* kb = qb + NTOK * E_;
  _Float16* vb = kb + NTOK * E_;
  _Float16* ab = vb + NTOK * E_;
  _Float16* vtb = ab + NTOK * E_;           // V transposed [bh, d, s]

  dim3 gg(E_ / 128, (unsigned)(NTOK / 128)), bg(256);
  proj_gemm_kernel<false><<<gg, bg, 0, stream>>>(query, Wq, bq, nullptr, qb);
  proj_gemm_kernel<true ><<<gg, bg, 0, stream>>>(key,   Wk, bk, ts,      kb);
  proj_gemm_kernel<false><<<gg, bg, 0, stream>>>(value, Wv, bv, nullptr, vb);

  transpose_v_kernel<<<dim3(S_ / 64, B_ * H_), bg, 0, stream>>>(vb, vtb);

  const size_t smem =
      (size_t)(32 * SC_STRIDE + 32 * OA_STRIDE + 32) * sizeof(float); // ~271 KB
  (void)hipFuncSetAttribute((const void*)attn_kernel,
                            hipFuncAttributeMaxDynamicSharedMemorySize,
                            (int)smem);
  attn_kernel<<<dim3(B_ * H_ * (S_ / 32)), bg, smem, stream>>>(qb, kb, vtb, ab);

  out_gemm_kernel<<<gg, bg, 0, stream>>>(ab, Wo, bo, out);
}
